// RandomFourrierFeaturesModel_75041668596244
// MI455X (gfx1250) — matrix operand order repaired and checked by reading, not compiled, not run on hardware
//
#include <hip/hip_runtime.h>

// Problem constants (fixed by the reference setup)
#define NELEM   4
#define NMOL    64
#define NATOMS  16
#define NA      1024          // nmol*natoms
#define REP_DIM 256
#define NPCAS   128           // K dimension of the big GEMM
#define DDIM    131072
#define NSLOTS  1152          // 1024 atoms + per-element pad to 16, rounded up
#define NORM_F  0.00390625f   // sqrt(2/131072) == 1/256

typedef __attribute__((ext_vector_type(16))) __bf16 v16bf;
typedef __attribute__((ext_vector_type(8)))  float  v8f;

__device__ __forceinline__ unsigned short f32_to_bf16_rne(float f) {
    union { float f; unsigned u; } x; x.f = f;
    unsigned r = x.u + 0x7FFFu + ((x.u >> 16) & 1u);
    return (unsigned short)(r >> 16);
}

// VALU 16-lane butterfly sum via DPP: masks {1,2,7,15} span GF(2)^4, all
// permutations stay inside each 16-lane row (wave32 = two independent rows).
template <int CTRL>
__device__ __forceinline__ float dpp_add(float v) {
    int p = __builtin_amdgcn_update_dpp(0, __float_as_int(v), CTRL, 0xF, 0xF, true);
    return v + __int_as_float(p);
}
__device__ __forceinline__ float rowsum16(float v) {
    v = dpp_add<0xB1>(v);   // quad_perm [1,0,3,2]  : xor 1
    v = dpp_add<0x4E>(v);   // quad_perm [2,3,0,1]  : xor 2
    v = dpp_add<0x141>(v);  // row_half_mirror      : xor 7
    v = dpp_add<0x140>(v);  // row_mirror           : xor 15
    return v;
}

// ---------------------------------------------------------------------------
// Kernel 1: deterministic grouping of atoms by element + tile table + zeroing.
// ---------------------------------------------------------------------------
__global__ void setup_kernel(const int* __restrict__ Zidx,
                             int* __restrict__ meta,
                             int* __restrict__ slotOf,
                             int* __restrict__ perm,
                             int* __restrict__ tileTable,
                             unsigned short* __restrict__ subBuf,
                             float* __restrict__ out) {
    __shared__ int sZ[NA];
    __shared__ int sPos[NA];
    __shared__ int sStart[NELEM];
    const int tid = threadIdx.x;            // 256 threads

    if (tid < NMOL) out[tid] = 0.0f;
    for (int i = tid; i < NSLOTS; i += 256) perm[i] = -1;
    uint4 z4 = {0u, 0u, 0u, 0u};
    uint4* sb = (uint4*)subBuf;
    for (int i = tid; i < NSLOTS * NPCAS * 2 / 16; i += 256) sb[i] = z4;
    for (int a = tid; a < NA; a += 256) sZ[a] = Zidx[a];
    __syncthreads();

    for (int a = tid; a < NA; a += 256) {
        int z = sZ[a], p = 0;
        for (int a2 = 0; a2 < a; ++a2) p += (sZ[a2] == z);
        sPos[a] = p;
    }
    __syncthreads();

    if (tid == 0) {
        int cnt[NELEM] = {0, 0, 0, 0};
        for (int a = 0; a < NA; ++a) cnt[sZ[a]]++;
        int s = 0, nt = 0;
        for (int e = 0; e < NELEM; ++e) {
            sStart[e] = s;
            int tiles = (cnt[e] + 15) >> 4;
            for (int t = 0; t < tiles; ++t) {
                tileTable[2 * nt]     = e;
                tileTable[2 * nt + 1] = s + t * 16;
                nt++;
            }
            s += tiles * 16;
        }
        meta[0] = nt;
    }
    __syncthreads();

    for (int a = tid; a < NA; a += 256) {
        int slot = sStart[sZ[a]] + sPos[a];
        perm[slot] = a;
        slotOf[a]  = slot;
    }
}

// ---------------------------------------------------------------------------
// Kernel 2: sub[a,p] = rep[a,:] @ reductors[z[a],:,p], bf16, grouped order.
// ---------------------------------------------------------------------------
__global__ void project_kernel(const float* __restrict__ rep,
                               const float* __restrict__ red,
                               const int* __restrict__ Zidx,
                               const int* __restrict__ slotOf,
                               unsigned short* __restrict__ subBuf) {
    __shared__ float srep[REP_DIM];
    const int a = blockIdx.x;               // 1024 blocks, one atom each
    const int p = threadIdx.x;              // 128 threads, one PCA each
    srep[p]       = rep[a * REP_DIM + p];
    srep[p + 128] = rep[a * REP_DIM + p + 128];
    __syncthreads();
    const int z = Zidx[a];
    const float* wr = red + (size_t)z * REP_DIM * NPCAS + p;
    float acc = 0.0f;
#pragma unroll 4
    for (int r = 0; r < REP_DIM; ++r) acc += srep[r] * wr[(size_t)r * NPCAS];
    subBuf[(size_t)slotOf[a] * NPCAS + p] = f32_to_bf16_rne(acc);
}

// ---------------------------------------------------------------------------
// Kernel 3: fused WMMA pass. One wave owns 16 columns of D; W[e] slab held in
// registers (reloaded <=4x since tiles are element-sorted -> W streamed from
// HBM exactly once). Unroll-by-2 software pipeline: tile t+1's A-loads are in
// flight under tile t's WMMA + cos/alpha/mol epilogue.
// ---------------------------------------------------------------------------
struct ATile { union { v16bf v; uint4 q[2]; } a[4]; };

__global__ void __launch_bounds__(256)
rff_main_kernel(const float* __restrict__ W,
                const float* __restrict__ bvec,
                const float* __restrict__ alpha,
                const int* __restrict__ perm,
                const int* __restrict__ tileTable,
                const int* __restrict__ meta,
                const unsigned short* __restrict__ subBuf,
                float* __restrict__ out) {
    __shared__ float molsum[NMOL];
    const int tid = threadIdx.x;
    if (tid < NMOL) molsum[tid] = 0.0f;
    __syncthreads();

    const int wave  = tid >> 5;
    const int lane  = tid & 31;
    const int laneN = lane & 15;            // column within 16-wide tile
    const int hi    = lane >> 4;            // half-wave select
    const int col   = (blockIdx.x * 8 + wave) * 16 + laneN;
    const float alphaL = alpha[col];
    const int numTiles = meta[0];

    v16bf B0{}, B1{}, B2{}, B3{};
    float beCur = 0.0f;
    int curE = -1;

    // A layout: lane = M row; per k-step, halves {0..7} = K base..base+7,
    // halves {8..15} = K base+16..base+23, base = k*32 + hi*8.
    auto loadA = [&](int t, ATile& A) {
        const int base = tileTable[2 * t + 1];
        const unsigned short* ap = subBuf + (size_t)(base + laneN) * NPCAS + hi * 8;
#pragma unroll
        for (int k = 0; k < 4; ++k) {
            A.a[k].q[0] = *(const uint4*)(ap + k * 32);
            A.a[k].q[1] = *(const uint4*)(ap + k * 32 + 16);
        }
    };

    auto process = [&](int t, ATile& A) {
        const int e    = tileTable[2 * t];
        const int base = tileTable[2 * t + 1];

        if (e != curE) {                    // <=4 reloads per wave (sorted tiles)
            curE = e;
            beCur = bvec[(size_t)e * DDIM + col];
            const float* wp = W + (size_t)e * NPCAS * (size_t)DDIM + col;
#pragma unroll
            for (int k = 0; k < 4; ++k) {
                v16bf bk{};
#pragma unroll
                for (int j = 0; j < 16; ++j) {
                    // B layout: lane = N, halves j<8 hold K = k*32 + hi*8 + j, j>=8 hold K = k*32 + 16 + hi*8 + (j-8)
                    const int K = k * 32 + hi * 8 + j + (j >= 8 ? 8 : 0);
                    union { unsigned short s; __bf16 h; } y;
                    y.s = f32_to_bf16_rne(wp[(size_t)K * DDIM]);
                    bk[j] = y.h;
                }
                if      (k == 0) B0 = bk;
                else if (k == 1) B1 = bk;
                else if (k == 2) B2 = bk;
                else             B3 = bk;
            }
        }

        v8f acc = {0.f, 0.f, 0.f, 0.f, 0.f, 0.f, 0.f, 0.f};
        acc = __builtin_amdgcn_wmma_f32_16x16x32_bf16(false, A.a[0].v, false, B0, (short)0, acc, false, false);
        acc = __builtin_amdgcn_wmma_f32_16x16x32_bf16(false, A.a[1].v, false, B1, (short)0, acc, false, false);
        acc = __builtin_amdgcn_wmma_f32_16x16x32_bf16(false, A.a[2].v, false, B2, (short)0, acc, false, false);
        acc = __builtin_amdgcn_wmma_f32_16x16x32_bf16(false, A.a[3].v, false, B3, (short)0, acc, false, false);

        // D rows M = r + hi*8 live in acc[r]. Tile's atom ids in 2 vector loads.
        const int4 pa = *(const int4*)(perm + base + hi * 8);
        const int4 pb = *(const int4*)(perm + base + hi * 8 + 4);
        const int atoms[8] = {pa.x, pa.y, pa.z, pa.w, pb.x, pb.y, pb.z, pb.w};
#pragma unroll
        for (int r = 0; r < 8; ++r) {
            const int atom = atoms[r];
            const float wgt = (atom >= 0) ? alphaL : 0.0f;  // gate folded, no branch
            float v = __cosf(acc[r] + beCur) * wgt;
            v = rowsum16(v);                                 // pure-VALU DPP butterfly
            if (laneN == 0 && atom >= 0) atomicAdd(&molsum[atom >> 4], v);
        }
    };

    ATile Abuf0, Abuf1;
    if (numTiles > 0) loadA(0, Abuf0);
    int t = 0;
    for (; t + 1 < numTiles; t += 2) {
        loadA(t + 1, Abuf1);                 // in flight under process(t)
        process(t, Abuf0);
        if (t + 2 < numTiles) loadA(t + 2, Abuf0);
        process(t + 1, Abuf1);
    }
    if (t < numTiles) process(t, Abuf0);

    __syncthreads();
    if (tid < NMOL) atomicAdd(&out[tid], NORM_F * molsum[tid]);
}

// ---------------------------------------------------------------------------
extern "C" void kernel_launch(void* const* d_in, const int* in_sizes, int n_in,
                              void* d_out, int out_size, void* d_ws, size_t ws_size,
                              hipStream_t stream) {
    const float* rep   = (const float*)d_in[0];  // [64,16,256]
    const float* red   = (const float*)d_in[1];  // [4,256,128]
    const float* W     = (const float*)d_in[2];  // [4,128,131072]
    const float* bvec  = (const float*)d_in[3];  // [4,131072]
    const float* alpha = (const float*)d_in[4];  // [131072]
    const int*   Zidx  = (const int*)d_in[5];    // [64,16]
    float* out = (float*)d_out;                  // [64]

    char* ws = (char*)d_ws;
    int* meta      = (int*)ws;                                 //   64 B
    int* slotOf    = (int*)(ws + 64);                          // 4096 B
    int* perm      = (int*)(ws + 64 + 4096);                   // 4608 B
    int* tileTable = (int*)(ws + 64 + 4096 + 4608);            //  640 B
    unsigned short* subBuf = (unsigned short*)(ws + 64 + 4096 + 4608 + 640); // 294912 B

    setup_kernel<<<1, 256, 0, stream>>>(Zidx, meta, slotOf, perm, tileTable, subBuf, out);
    project_kernel<<<NA, 128, 0, stream>>>(rep, red, Zidx, slotOf, subBuf);
    rff_main_kernel<<<DDIM / 128, 256, 0, stream>>>(W, bvec, alpha, perm, tileTable,
                                                    meta, subBuf, out);
}
